// TernaryMNISTNet_35270271435508
// MI455X (gfx1250) — matrix operand order repaired and checked by reading, not compiled, not run on hardware
//
#include <hip/hip_runtime.h>
#include <hip/hip_fp16.h>

typedef __attribute__((ext_vector_type(16))) _Float16 v16h;
typedef __attribute__((ext_vector_type(8)))  float    v8f;
typedef __attribute__((ext_vector_type(4)))  float    v4f;

#define T_STEPS 100
#define BATCH   1024
#define D_IN    784
#define H_DIM   256
#define O_DIM   10
#define KPAD    800           // 784 padded up to 25 chunks of 32
#define NCHUNK  25
#define BETA    0.9f
#define THR     1.0f
#define M_TOTAL (T_STEPS * BATCH)   // 102400 rows of the big GEMM

// ---------------------------------------------------------------------------
// Kernel 1: W1 [256 x 784] fp32 -> fp16, zero-padded to K=800 per row.
// ---------------------------------------------------------------------------
__global__ __launch_bounds__(256)
void cvt_w1_kernel(const float* __restrict__ W1, _Float16* __restrict__ W1h) {
    int idx = blockIdx.x * 256 + threadIdx.x;      // 0 .. 256*800-1 (grid = 800)
    int row = idx / KPAD;
    int col = idx - row * KPAD;
    float v = (col < D_IN) ? W1[row * D_IN + col] : 0.0f;
    W1h[idx] = (_Float16)v;
}

// ---------------------------------------------------------------------------
// Kernel 2: cur1[m, h] = sum_k x[m, k] * W1[h, k]   (M=102400, N=256, K=784)
// 8 waves / WG; each wave computes a 16x256 strip with 16 f16 WMMA tiles.
//  - W1h chunks staged into LDS in B-fragment layout, double buffered (LDS)
//  - B fragments double-buffered 2-deep in registers (hide ds latency)
//  - A-chunk global loads software-pipelined one chunk ahead (hide HBM latency)
// ---------------------------------------------------------------------------
__global__ __launch_bounds__(256)
void gemm1_kernel(const float* __restrict__ x,
                  const _Float16* __restrict__ W1h,
                  float* __restrict__ cur1) {
    // 2 buffers * 16 tiles * (32 lanes * 16 halves) = 16384 halves = 32 KB
    __shared__ alignas(64) _Float16 ldsB[2 * 16 * 512];

    const int tid  = threadIdx.x;
    const int lane = tid & 31;
    const int wave = tid >> 5;
    const size_t m0 = (size_t)blockIdx.x * 128 + (size_t)wave * 16;

    v8f acc[16];
    #pragma unroll
    for (int j = 0; j < 16; ++j)
        #pragma unroll
        for (int r = 0; r < 8; ++r) acc[j][r] = 0.0f;

    // --- stage W1h chunk k0 into LDS buffer `buf` in B-fragment layout ----
    // B frag (16x16x32 f16): lane l<16 -> N=l, K=k0..k0+7 & k0+16..23 ; lane>=16 -> N=l-16,
    // K=k0+8..15 & k0+24..31.  Thread `tid` stages W1 row n=tid (64 contiguous bytes).
    auto stage = [&](int buf, int k0) {
        const v16h* src = (const v16h*)(W1h + (size_t)tid * KPAD + k0);
        v16h s0 = src[0];                       // K k0..k0+15
        v16h s1 = src[1];                       // K k0+16..k0+31
        _Float16* dst = ldsB + buf * 8192 + (tid >> 4) * 512;
        int rn = tid & 15;
        *(v16h*)(dst + rn * 16)        = __builtin_shufflevector(s0, s1, 0, 1, 2, 3, 4, 5, 6, 7, 16, 17, 18, 19, 20, 21, 22, 23);    // lane rn
        *(v16h*)(dst + (rn + 16) * 16) = __builtin_shufflevector(s0, s1, 8, 9, 10, 11, 12, 13, 14, 15, 24, 25, 26, 27, 28, 29, 30, 31);    // lane rn+16
    };

    // --- raw fp32 A loads for this wave's 16 rows, K chunk k0 --------------
    // A frag (16x32 f16): lane&15 = M row; lanes<16: K = k0+0..7 & k0+16..23,
    // lanes>=16: K = k0+8..15 & k0+24..31.
    const float* xrow = x + (m0 + (size_t)(lane & 15)) * (size_t)D_IN;
    const int klsel = (lane & 16) ? 8 : 0;

    struct ARaw { v4f a0, a1, b0, b1; };
    auto loadAraw = [&](int k0) -> ARaw {
        const int klo = k0 + klsel;                   // always < 784
        const int khi = klo + 16;                     // may be >= 784 (tail)
        const bool ok = (khi < D_IN);
        const v4f* p0 = (const v4f*)(xrow + klo);
        const v4f* p1 = (const v4f*)(xrow + (ok ? khi : 0));   // safe addr
        ARaw r;
        r.a0 = p0[0]; r.a1 = p0[1];
        r.b0 = p1[0]; r.b1 = p1[1];
        if (!ok) {
            v4f z = {0.f, 0.f, 0.f, 0.f};
            r.b0 = z; r.b1 = z;
        }
        return r;
    };
    auto cvtA = [&](const ARaw& r) -> v16h {
        v16h a;
        #pragma unroll
        for (int i = 0; i < 4; ++i) {
            a[i]      = (_Float16)r.a0[i];
            a[i + 4]  = (_Float16)r.a1[i];
            a[i + 8]  = (_Float16)r.b0[i];
            a[i + 12] = (_Float16)r.b1[i];
        }
        return a;
    };

    stage(0, 0);
    ARaw araw = loadAraw(0);

    for (int kk = 0; kk < NCHUNK; ++kk) {
        const int k0 = kk * 32;
        __syncthreads();
        if (kk + 1 < NCHUNK) stage((kk + 1) & 1, k0 + 32);   // overlap staging

        v16h a = cvtA(araw);
        ARaw anext;
        if (kk + 1 < NCHUNK) anext = loadAraw(k0 + 32);      // prefetch x chunk

        const _Float16* base = ldsB + (kk & 1) * 8192 + lane * 16;
        // 2-deep register pipeline of B fragments over the 16 WMMAs
        v16h b0 = *(const v16h*)(base);
        v16h b1 = *(const v16h*)(base + 512);
        #pragma unroll
        for (int j = 0; j < 16; ++j) {
            v16h b2;
            if (j + 2 < 16) b2 = *(const v16h*)(base + (j + 2) * 512);
            acc[j] = __builtin_amdgcn_wmma_f32_16x16x32_f16(
                false, a, false, b0, (short)0, acc[j], false, false);
            b0 = b1;
            if (j + 2 < 16) b1 = b2;
        }
        if (kk + 1 < NCHUNK) araw = anext;
    }

    // --- epilogue: C/D layout: lanes 0-15 -> M=r, N=lane; 16-31 -> M=8+r ---
    const int    nlo  = lane & 15;
    const size_t mrow = m0 + ((lane & 16) ? 8 : 0);
    #pragma unroll
    for (int j = 0; j < 16; ++j) {
        size_t cbase = mrow * H_DIM + (size_t)(j * 16 + nlo);
        #pragma unroll
        for (int r = 0; r < 8; ++r)
            cur1[cbase + (size_t)r * H_DIM] = acc[j][r];
    }
}

// ---------------------------------------------------------------------------
// Kernel 3: sequential T-scan. 1 block = 1 batch element, thread h owns
// mem1[h]; layer-2 reduced with wave32 shuffles + tiny LDS combine.
// ---------------------------------------------------------------------------
__global__ __launch_bounds__(256)
void snn_scan_kernel(const float* __restrict__ cur1,
                     const float* __restrict__ b1,
                     const float* __restrict__ W2,
                     const float* __restrict__ b2,
                     float* __restrict__ out) {
    __shared__ float wsum[8 * O_DIM];

    const int h    = threadIdx.x;        // hidden unit 0..255
    const int b    = blockIdx.x;         // batch element 0..1023
    const int lane = h & 31;
    const int wave = h >> 5;

    float w2r[O_DIM];
    #pragma unroll
    for (int o = 0; o < O_DIM; ++o) w2r[o] = W2[o * H_DIM + h];
    const float b1h = b1[h];
    const float b2o = (h < O_DIM) ? b2[h] : 0.0f;

    float mem1 = 0.0f;
    float mem2 = 0.0f;
    float accO = 0.0f;

    for (int t = 0; t < T_STEPS; ++t) {
        float cur = cur1[((size_t)t * BATCH + b) * H_DIM + h];
        mem1 = BETA * mem1 + cur + b1h;
        float spk = (mem1 > THR ? 1.0f : 0.0f) - (mem1 < -THR ? 1.0f : 0.0f);
        mem1 -= spk * THR;

        // cur2[o] = sum_h spk[h] * W2[o,h] : per-wave shuffle tree, then LDS
        #pragma unroll
        for (int o = 0; o < O_DIM; ++o) {
            float v = spk * w2r[o];
            #pragma unroll
            for (int off = 16; off > 0; off >>= 1)
                v += __shfl_xor(v, off, 32);
            if (lane == 0) wsum[wave * O_DIM + o] = v;
        }
        __syncthreads();
        if (h < O_DIM) {
            float c2 = b2o;
            #pragma unroll
            for (int w = 0; w < 8; ++w) c2 += wsum[w * O_DIM + h];
            mem2 = BETA * mem2 + c2;
            float s2 = (mem2 > THR ? 1.0f : 0.0f) - (mem2 < -THR ? 1.0f : 0.0f);
            mem2 -= s2 * THR;
            accO += s2;
        }
        __syncthreads();
    }
    if (h < O_DIM) out[b * O_DIM + h] = accO * (1.0f / (float)T_STEPS);
}

// ---------------------------------------------------------------------------
extern "C" void kernel_launch(void* const* d_in, const int* in_sizes, int n_in,
                              void* d_out, int out_size, void* d_ws, size_t ws_size,
                              hipStream_t stream) {
    const float* x  = (const float*)d_in[0];   // [100,1024,784]
    const float* W1 = (const float*)d_in[1];   // [256,784]
    const float* b1 = (const float*)d_in[2];   // [256]
    const float* W2 = (const float*)d_in[3];   // [10,256]
    const float* b2 = (const float*)d_in[4];   // [10]
    float* out = (float*)d_out;                // [1024,10]

    // workspace: cur1 (102400*256 f32 = 100 MB), then W1h fp16 padded (400 KB)
    float*    cur1 = (float*)d_ws;
    _Float16* W1h  = (_Float16*)((char*)d_ws + (size_t)M_TOTAL * H_DIM * sizeof(float));

    cvt_w1_kernel  <<<dim3((256 * KPAD) / 256), dim3(256), 0, stream>>>(W1, W1h);
    gemm1_kernel   <<<dim3(M_TOTAL / 128),      dim3(256), 0, stream>>>(x, W1h, cur1);
    snn_scan_kernel<<<dim3(BATCH),              dim3(256), 0, stream>>>(cur1, b1, W2, b2, out);
}
